// HyperAttention_29497835389546
// MI455X (gfx1250) — matrix operand order repaired and checked by reading, not compiled, not run on hardware
//
#include <hip/hip_runtime.h>

// ---------------------------------------------------------------------------
// Problem constants (from reference setup_inputs)
// ---------------------------------------------------------------------------
#define BSZ   1024
#define NROW  64
#define EDIM  1024
#define DDIM  512
#define SCALE 0.03125f        // E^-0.5 = 1/32
#define EPSV  1e-5f

typedef __attribute__((ext_vector_type(8)))  float  v8f;
typedef __attribute__((ext_vector_type(8)))  __bf16 v8bf;
typedef __attribute__((ext_vector_type(16))) __bf16 v16bf;

// physical row permutation: logical row m -> (m&15)*4 + (m>>4)
// => lane lr's four mt-rows {lr, lr+16, lr+32, lr+48} are adjacent in LDS.
__device__ __forceinline__ int prow(int m) { return (m & 15) * 4 + (m >> 4); }

// ---------------------------------------------------------------------------
// Kernel 1: per-batch LN(context) -> K = nk@kW.T + kb, V = nk@vW.T + vb
// ---------------------------------------------------------------------------
__global__ void ha_prep_kv(const float* __restrict__ ctx,
                           const float* __restrict__ kW, const float* __restrict__ kb,
                           const float* __restrict__ vW, const float* __restrict__ vb,
                           const float* __restrict__ lnk_g, const float* __restrict__ lnk_b,
                           float* __restrict__ Kv, float* __restrict__ Vv) {
    __shared__ float nk[DDIM];
    __shared__ float r1[256], r2[256];
    const int b = blockIdx.x;
    const int t = threadIdx.x;

    const float x0 = ctx[b * DDIM + 2 * t];
    const float x1 = ctx[b * DDIM + 2 * t + 1];
    r1[t] = x0 + x1;
    r2[t] = x0 * x0 + x1 * x1;
    __syncthreads();
    for (int off = 128; off > 0; off >>= 1) {
        if (t < off) { r1[t] += r1[t + off]; r2[t] += r2[t + off]; }
        __syncthreads();
    }
    const float mu = r1[0] * (1.0f / DDIM);
    const float var = r2[0] * (1.0f / DDIM) - mu * mu;
    const float rs = rsqrtf(var + EPSV);
    nk[2 * t]     = (x0 - mu) * rs * lnk_g[2 * t]     + lnk_b[2 * t];
    nk[2 * t + 1] = (x1 - mu) * rs * lnk_g[2 * t + 1] + lnk_b[2 * t + 1];
    __syncthreads();

    for (int j = 0; j < 4; ++j) {
        const int e = j * 256 + t;
        float ak = kb[e], av = vb[e];
        const float* kr = kW + (size_t)e * DDIM;
        const float* vr = vW + (size_t)e * DDIM;
        #pragma unroll 4
        for (int d = 0; d < DDIM; ++d) {
            const float n = nk[d];
            ak += n * kr[d];
            av += n * vr[d];
        }
        Kv[(size_t)b * EDIM + e] = ak;
        Vv[(size_t)b * EDIM + e] = av;
    }
}

// ---------------------------------------------------------------------------
// Kernel 2/3: small fp32 tiled GEMM, C = A @ B (B KxN, or NxK when TRANSB).
// Used for WQ = K @ qW  and  OV = V @ oW.T  (M=N=K=1024).
// ---------------------------------------------------------------------------
template <bool TRANSB>
__global__ void ha_gemm32(const float* __restrict__ A, const float* __restrict__ Bm,
                          float* __restrict__ C, int M, int N, int K) {
    __shared__ float As[32][33];
    __shared__ float Bs[32][33];
    const int t  = threadIdx.x;
    const int tx = t & 15, ty = t >> 4;
    const int row0 = blockIdx.y * 32, col0 = blockIdx.x * 32;
    float c00 = 0.f, c01 = 0.f, c10 = 0.f, c11 = 0.f;

    for (int k0 = 0; k0 < K; k0 += 32) {
        for (int i = t; i < 32 * 32; i += 256) {
            const int r = i >> 5, c = i & 31;
            As[r][c] = A[(size_t)(row0 + r) * K + k0 + c];
            Bs[r][c] = TRANSB ? Bm[(size_t)(col0 + c) * K + k0 + r]
                              : Bm[(size_t)(k0 + r) * N + col0 + c];
        }
        __syncthreads();
        #pragma unroll
        for (int kk = 0; kk < 32; ++kk) {
            const float a0 = As[2 * ty][kk], a1 = As[2 * ty + 1][kk];
            const float b0 = Bs[kk][2 * tx], b1 = Bs[kk][2 * tx + 1];
            c00 += a0 * b0; c01 += a0 * b1;
            c10 += a1 * b0; c11 += a1 * b1;
        }
        __syncthreads();
    }
    C[(size_t)(row0 + 2 * ty) * N + col0 + 2 * tx]         = c00;
    C[(size_t)(row0 + 2 * ty) * N + col0 + 2 * tx + 1]     = c01;
    C[(size_t)(row0 + 2 * ty + 1) * N + col0 + 2 * tx]     = c10;
    C[(size_t)(row0 + 2 * ty + 1) * N + col0 + 2 * tx + 1] = c11;
}

// ---------------------------------------------------------------------------
// Kernel 4: per-batch fold of LN-q affine + biases into score constants.
// ---------------------------------------------------------------------------
__global__ void ha_batch_scalars(const float* __restrict__ WQ, const float* __restrict__ Kv,
                                 const float* __restrict__ lnq_g, const float* __restrict__ lnq_b,
                                 const float* __restrict__ qb,
                                 float* __restrict__ GWQ, float* __restrict__ sumg,
                                 float* __restrict__ sc0) {
    __shared__ float r1[256], r2[256], r3[256];
    const int b = blockIdx.x;
    const int t = threadIdx.x;
    float sg = 0.f, sb = 0.f, qk = 0.f;
    for (int i = t; i < EDIM; i += 256) {
        const float wq = WQ[(size_t)b * EDIM + i];
        const float gw = lnq_g[i] * wq;
        GWQ[(size_t)b * EDIM + i] = gw;
        sg += gw;
        sb += lnq_b[i] * wq;
        qk += qb[i] * Kv[(size_t)b * EDIM + i];
    }
    r1[t] = sg; r2[t] = sb; r3[t] = qk;
    __syncthreads();
    for (int off = 128; off > 0; off >>= 1) {
        if (t < off) { r1[t] += r1[t + off]; r2[t] += r2[t + off]; r3[t] += r3[t + off]; }
        __syncthreads();
    }
    if (t == 0) { sumg[b] = r1[0]; sc0[b] = r2[0] + r3[0]; }
}

// ---------------------------------------------------------------------------
// Kernel 5: oW (fp32, ExE) -> bf16
// ---------------------------------------------------------------------------
__global__ void ha_cvt_bf16(const float* __restrict__ src, __bf16* __restrict__ dst, int n) {
    const int i = blockIdx.x * blockDim.x + threadIdx.x;
    if (i < n) dst[i] = (__bf16)src[i];
}

// ---------------------------------------------------------------------------
// Kernel 6 (main)
// ---------------------------------------------------------------------------
#define ROWP 1026   // 1024 + 2 bf16 pad: 513 dwords/row == 1 (mod 16)
                    // -> 4-row lane stride == 4 banks (mod 64): optimal 2-pass LDS reads
#define SM_TILE_B   (NROW * ROWP * 2)               // 131328
#define SM_GWQ_B    (SM_TILE_B)
#define SM_OV_B     (SM_GWQ_B + EDIM * 4)
#define SM_OB_B     (SM_OV_B + EDIM * 4)
#define SM_ATTN_B   (SM_OB_B + EDIM * 4)
#define SM_RED_B    (SM_ATTN_B + NROW * 4)
#define SM_TOTAL_B  (SM_RED_B + NROW * 4 * 3 * 4)   // 146944 bytes (< 320KB WGP LDS)

__global__ void ha_main(const float* __restrict__ role,
                        const __bf16* __restrict__ owb,
                        const float* __restrict__ GWQ, const float* __restrict__ OV,
                        const float* __restrict__ obias,
                        const float* __restrict__ sumg, const float* __restrict__ sc0,
                        float* __restrict__ out) {
    extern __shared__ char smem[];
    __bf16* tile  = (__bf16*)smem;
    float* gwq_l  = (float*)(smem + SM_GWQ_B);
    float* ov_l   = (float*)(smem + SM_OV_B);
    float* ob_l   = (float*)(smem + SM_OB_B);
    float* attn_l = (float*)(smem + SM_ATTN_B);
    float* red    = (float*)(smem + SM_RED_B);

    const int b = blockIdx.x;
    const int t = threadIdx.x;

    // ---- Phase A0: async global->LDS copy of per-batch vectors (pure byte
    // moves, ASYNCcnt-tracked; overlaps with the bf16 tile conversion below).
    {
        const float* gsrc[3] = { GWQ + (size_t)b * EDIM, OV + (size_t)b * EDIM, obias };
        float* const  ldst[3] = { gwq_l, ov_l, ob_l };
        #pragma unroll
        for (int j = 0; j < 3; ++j) {
            const float* src = gsrc[j] + t * 4;                 // 16B per lane
            const unsigned int dst = (unsigned int)(uintptr_t)(ldst[j] + t * 4);
            asm volatile("global_load_async_to_lds_b128 %0, %1, off"
                         :: "v"(dst), "v"(src) : "memory");
        }
    }

    // ---- Phase A1: coalesced stage-in of role tile -> bf16 LDS (permuted rows)
    const float4* role4 = (const float4*)(role + (size_t)b * NROW * EDIM);
    for (int vid = t; vid < NROW * EDIM / 4; vid += 256) {
        const int row = vid >> 8;           // 256 float4 per 1024-wide row
        const int col = (vid & 255) * 4;
        const float4 x = role4[vid];
        __bf16* dst = tile + prow(row) * ROWP + col;
        dst[0] = (__bf16)x.x; dst[1] = (__bf16)x.y;
        dst[2] = (__bf16)x.z; dst[3] = (__bf16)x.w;
    }
    asm volatile("s_wait_asynccnt 0x0" ::: "memory");
    __syncthreads();

    // ---- Phase B: per-row LN stats + score + tanh (from LDS)
    {
        const int row = t >> 2, seg = t & 3, base = seg * 256;
        float s = 0.f, ss = 0.f, dt = 0.f;
        const __bf16* rp = tile + prow(row) * ROWP;
        #pragma unroll 4
        for (int c = base; c < base + 256; ++c) {
            const float xv = (float)rp[c];
            s += xv; ss += xv * xv; dt += xv * gwq_l[c];
        }
        float* pr = red + (row * 4 + seg) * 3;
        pr[0] = s; pr[1] = ss; pr[2] = dt;
    }
    __syncthreads();
    if ((t & 3) == 0) {
        const int row = t >> 2;
        float s = 0.f, ss = 0.f, dt = 0.f;
        for (int j = 0; j < 4; ++j) {
            const float* pr = red + (row * 4 + j) * 3;
            s += pr[0]; ss += pr[1]; dt += pr[2];
        }
        const float mu  = s * (1.0f / EDIM);
        const float var = ss * (1.0f / EDIM) - mu * mu;
        const float rs  = rsqrtf(var + EPSV);
        const float score = (rs * (dt - mu * sumg[b]) + sc0[b]) * SCALE;
        attn_l[row] = tanhf(score);
    }
    __syncthreads();

    // ---- Phase C: WMMA GEMM role_tile(64x1024,bf16) @ oW^T, nt-outer /
    // mt-inner: each B fragment loaded once, reused by 4 WMMAs.  Permuted
    // tile rows keep all 8 DS loads per k-step as static offsets (< 8.3KB)
    // off one base register: no per-iteration address VALU, fewer hazard NOPs.
    const int wave = t >> 5;
    const int lane = t & 31;
    const int lr   = lane & 15;
    const int hi   = lane >> 4;          // 0: lanes 0-15, 1: lanes 16-31
    const int nbase = wave * 128;

    const __bf16* Abase = tile + (lr * 4) * ROWP + hi * 8;   // physical rows lr*4 + mt

    for (int nt = 0; nt < 8; ++nt) {
        const int col0 = nbase + nt * 16;
        // B operand: column n = col0+lr of oW^T == row n of oW (contiguous K)
        const __bf16* Bp = owb + (size_t)(col0 + lr) * EDIM + hi * 8;
        v8f acc[4] = {{}, {}, {}, {}};
        #pragma unroll 2
        for (int k0 = 0; k0 < EDIM; k0 += 32) {
            const v8bf b_lo = *(const v8bf*)(Bp + k0);
            const v8bf b_hi = *(const v8bf*)(Bp + k0 + 16);
            const v16bf bv = __builtin_shufflevector(b_lo, b_hi,
                0, 1, 2, 3, 4, 5, 6, 7, 8, 9, 10, 11, 12, 13, 14, 15);
            #pragma unroll
            for (int mt = 0; mt < 4; ++mt) {
                const __bf16* ap = Abase + mt * ROWP + k0;
                const v8bf a_lo = *(const v8bf*)(ap);         // K {0..7}/{8..15}
                const v8bf a_hi = *(const v8bf*)(ap + 16);    // K {16..23}/{24..31}
                const v16bf av = __builtin_shufflevector(a_lo, a_hi,
                    0, 1, 2, 3, 4, 5, 6, 7, 8, 9, 10, 11, 12, 13, 14, 15);
                acc[mt] = __builtin_amdgcn_wmma_f32_16x16x32_bf16(
                    false, av, false, bv, (short)0, acc[mt], false, false);
            }
        }
        // epilogue: D[m][n] += attn[m]*ov[n] + ob[n]
        const int n_e  = col0 + lr;
        const float ovv = ov_l[n_e];
        const float obb = ob_l[n_e];
        #pragma unroll
        for (int mt = 0; mt < 4; ++mt) {
            float* op = out + ((size_t)b * NROW + mt * 16 + hi * 8) * EDIM + n_e;
            #pragma unroll
            for (int vg = 0; vg < 8; ++vg) {
                const int m_e = mt * 16 + vg + hi * 8;
                op[(size_t)vg * EDIM] = acc[mt][vg] + attn_l[m_e] * ovv + obb;
            }
        }
    }
}

// ---------------------------------------------------------------------------
// Host launch
// ---------------------------------------------------------------------------
extern "C" void kernel_launch(void* const* d_in, const int* in_sizes, int n_in,
                              void* d_out, int out_size, void* d_ws, size_t ws_size,
                              hipStream_t stream) {
    const float* role  = (const float*)d_in[0];
    const float* ctx   = (const float*)d_in[1];
    const float* qW    = (const float*)d_in[2];
    const float* qb    = (const float*)d_in[3];
    const float* kW    = (const float*)d_in[4];
    const float* kb    = (const float*)d_in[5];
    const float* vW    = (const float*)d_in[6];
    const float* vb    = (const float*)d_in[7];
    const float* oW    = (const float*)d_in[8];
    const float* ob    = (const float*)d_in[9];
    const float* lnq_g = (const float*)d_in[10];
    const float* lnq_b = (const float*)d_in[11];
    const float* lnk_g = (const float*)d_in[12];
    const float* lnk_b = (const float*)d_in[13];
    float* outp = (float*)d_out;

    // workspace layout
    char* ws = (char*)d_ws;
    const size_t BE = (size_t)BSZ * EDIM * sizeof(float);   // 4 MB
    float*  Kv   = (float*)(ws);
    float*  Vv   = (float*)(ws + BE);
    float*  WQ   = (float*)(ws + 2 * BE);
    float*  OV   = (float*)(ws + 3 * BE);
    float*  GWQ  = (float*)(ws + 4 * BE);
    float*  sumg = (float*)(ws + 5 * BE);
    float*  sc0  = (float*)(ws + 5 * BE + BSZ * sizeof(float));
    __bf16* owb  = (__bf16*)(ws + 5 * BE + 2 * BSZ * sizeof(float));

    ha_prep_kv<<<BSZ, 256, 0, stream>>>(ctx, kW, kb, vW, vb, lnk_g, lnk_b, Kv, Vv);
    ha_gemm32<false><<<dim3(EDIM / 32, BSZ / 32), 256, 0, stream>>>(Kv, qW, WQ, BSZ, EDIM, EDIM);
    ha_gemm32<true><<<dim3(EDIM / 32, BSZ / 32), 256, 0, stream>>>(Vv, oW, OV, BSZ, EDIM, EDIM);
    ha_batch_scalars<<<BSZ, 256, 0, stream>>>(WQ, Kv, lnq_g, lnq_b, qb, GWQ, sumg, sc0);
    ha_cvt_bf16<<<(EDIM * EDIM + 255) / 256, 256, 0, stream>>>(oW, owb, EDIM * EDIM);
    ha_main<<<BSZ, 256, SM_TOTAL_B, stream>>>(role, owb, GWQ, OV, ob, sumg, sc0, outp);
}
